// AAUpsample1d_22162031247934
// MI455X (gfx1250) — matrix operand order repaired and checked by reading, not compiled, not run on hardware
//
#include <hip/hip_runtime.h>
#include <hip/hip_bf16.h>

typedef _Float16 v16h __attribute__((ext_vector_type(16)));
typedef _Float16 v8h  __attribute__((ext_vector_type(8)));
typedef float    v8f  __attribute__((ext_vector_type(8)));
typedef int      v4i  __attribute__((ext_vector_type(4)));

#define DIMC 512
#define LIN  4096
#define LOUT 8192
#define NBATCH 8

#define BM 64
#define BN 128
#define BK 32
#define APAD 40   // LDS row stride in halves (80B: 16B-aligned, conflict-free)

#define CAT16(lo,hi) __builtin_shufflevector(lo, hi, 0,1,2,3,4,5,6,7,8,9,10,11,12,13,14,15)

// ---- async global->LDS path (gfx1250 GLOBAL_LOAD_ASYNC_TO_LDS_B128), guarded ----
#if defined(__has_builtin)
#if __has_builtin(__builtin_amdgcn_global_load_async_to_lds_b128)
#define HAVE_ASYNC_LDS 1
#endif
#endif

#ifdef HAVE_ASYNC_LDS
typedef __attribute__((address_space(1))) v4i* gv4i_p;   // global int4*
typedef __attribute__((address_space(3))) v4i* lv4i_p;   // LDS int4*
#define ASYNC_CP_B128(lds_dst, glob_src)                                         \
    __builtin_amdgcn_global_load_async_to_lds_b128(                              \
        (gv4i_p)(glob_src), (lv4i_p)(lds_dst), 0, 0)
#if __has_builtin(__builtin_amdgcn_s_wait_asynccnt)
#define WAIT_ASYNC() __builtin_amdgcn_s_wait_asynccnt(0)
#else
#define WAIT_ASYNC() asm volatile("s_wait_asynccnt 0x0" ::: "memory")
#endif
#else
#define ASYNC_CP_B128(lds_dst, glob_src)                                         \
    (*(uint4*)(lds_dst) = *(const uint4*)(glob_src))
#define WAIT_ASYNC() ((void)0)
#endif

// ---------------------------------------------------------------------------
// Weight folding: A_k[e][c] = sum_d proj_w[e][d] * conv_w[d][c][k]  (f16 out)
// ---------------------------------------------------------------------------
__global__ __launch_bounds__(256)
void prep_Afold(const float* __restrict__ proj_w,
                const float* __restrict__ conv_w,
                _Float16* __restrict__ A0,
                _Float16* __restrict__ A1,
                _Float16* __restrict__ A2) {
    __shared__ float sp[16][17];
    __shared__ float sw[3][16][17];
    const int tx = threadIdx.x, ty = threadIdx.y;
    const int e0 = blockIdx.y * 16, c0 = blockIdx.x * 16;
    float acc0 = 0.f, acc1 = 0.f, acc2 = 0.f;
    for (int d0 = 0; d0 < DIMC; d0 += 16) {
        sp[ty][tx] = proj_w[(size_t)(e0 + ty) * DIMC + d0 + tx];
        const float* wp = conv_w + (size_t)(d0 + ty) * DIMC * 3 + (c0 + tx) * 3;
        sw[0][ty][tx] = wp[0];
        sw[1][ty][tx] = wp[1];
        sw[2][ty][tx] = wp[2];
        __syncthreads();
#pragma unroll
        for (int dd = 0; dd < 16; ++dd) {
            const float p = sp[ty][dd];
            acc0 += p * sw[0][dd][tx];
            acc1 += p * sw[1][dd][tx];
            acc2 += p * sw[2][dd][tx];
        }
        __syncthreads();
    }
    const size_t o = (size_t)(e0 + ty) * DIMC + c0 + tx;
    A0[o] = (_Float16)acc0;
    A1[o] = (_Float16)acc1;
    A2[o] = (_Float16)acc2;
}

// beff[e] = sum_d proj_w[e][d] * conv_b[d]
__global__ void prep_beff(const float* __restrict__ proj_w,
                          const float* __restrict__ conv_b,
                          float* __restrict__ beff) {
    const int e = blockIdx.x * blockDim.x + threadIdx.x;
    if (e >= DIMC) return;
    float acc = 0.f;
    for (int d = 0; d < DIMC; ++d) acc += proj_w[(size_t)e * DIMC + d] * conv_b[d];
    beff[e] = acc;
}

// ---------------------------------------------------------------------------
// Main WMMA GEMM:  z[b,e,2n]   = A1 x[:,n]            + beff
//                  z[b,e,2n+1] = A0 x[:,n] + A2 x[:,n+1] + beff   (f16 out)
// Block tile 64(M) x 128(N), 8 waves as 4(M) x 2(N), each wave 16x64.
// ---------------------------------------------------------------------------
__global__ __launch_bounds__(256)
void up_gemm(const float* __restrict__ x,
             const _Float16* __restrict__ A0,
             const _Float16* __restrict__ A1,
             const _Float16* __restrict__ A2,
             const float* __restrict__ beff,
             _Float16* __restrict__ zh) {
    __shared__ __align__(16) _Float16 sA[3][BM][APAD];      // A0/A1/A2 tiles
    __shared__ __align__(16) _Float16 sX[BN + 1][APAD];     // transposed X + halo

    const int tid  = threadIdx.x;
    const int b    = blockIdx.z;
    const int mb   = blockIdx.x * BM;   // M tiles in grid.x -> L2 reuse of X
    const int nb   = blockIdx.y * BN;

    const int wav  = tid >> 5;
    const int lane = tid & 31;
    const int mw   = (wav >> 1) * 16;   // wave M offset (0..48)
    const int nw   = (wav & 1) * 64;    // wave N offset (0/64)
    const int lm   = lane & 15;
    const int g    = lane >> 4;

    v8f accE0 = {}, accE1 = {}, accE2 = {}, accE3 = {};
    v8f accO0 = {}, accO1 = {}, accO2 = {}, accO3 = {};

    // A cooperative staging: 64 rows x 32 halves = one b128 per thread/matrix
    const int arow = tid >> 2;
    const int aseg = (tid & 3) * 8;

    for (int k0 = 0; k0 < DIMC; k0 += BK) {
        {   // stage A0/A1/A2 tiles (async DMA to LDS when available)
            const size_t go = (size_t)(mb + arow) * DIMC + k0 + aseg;
            ASYNC_CP_B128(&sA[0][arow][aseg], A0 + go);
            ASYNC_CP_B128(&sA[1][arow][aseg], A1 + go);
            ASYNC_CP_B128(&sA[2][arow][aseg], A2 + go);
        }
        // stage X tile transposed, f32 -> f16, with +1 halo column (zero at L)
        for (int i = tid; i < BK * (BN + 1); i += 256) {
            const int kk = i / (BN + 1);
            const int nn = i - kk * (BN + 1);
            const int ng = nb + nn;
            const float v = (ng < LIN)
                ? x[((size_t)b * DIMC + k0 + kk) * LIN + ng] : 0.f;
            sX[nn][kk] = (_Float16)v;
        }
        WAIT_ASYNC();
        __syncthreads();

        // A fragments: lane holds M=mw+lm, K = {g*8..g*8+7} U {g*8+16..g*8+23}
        v16h a0f, a1f, a2f;
        {
            const v8h l0 = *(const v8h*)&sA[0][mw + lm][g * 8];
            const v8h h0 = *(const v8h*)&sA[0][mw + lm][g * 8 + 16];
            a0f = CAT16(l0, h0);
            const v8h l1 = *(const v8h*)&sA[1][mw + lm][g * 8];
            const v8h h1 = *(const v8h*)&sA[1][mw + lm][g * 8 + 16];
            a1f = CAT16(l1, h1);
            const v8h l2 = *(const v8h*)&sA[2][mw + lm][g * 8];
            const v8h h2 = *(const v8h*)&sA[2][mw + lm][g * 8 + 16];
            a2f = CAT16(l2, h2);
        }

        // B fragments per 16-col subtile: lane holds N=col, K = {g*8..g*8+7} U {g*8+16..g*8+23}
#define DO_SUB(S, AE, AO)                                                          \
        {                                                                          \
            const int c0 = nw + (S) * 16 + lm;                                     \
            const v8h p  = *(const v8h*)&sX[c0][g * 8];                            \
            const v8h q  = *(const v8h*)&sX[c0][g * 8 + 16];                       \
            const v16h bf  = CAT16(p, q);                                          \
            const v8h ps = *(const v8h*)&sX[c0 + 1][g * 8];                        \
            const v8h qs = *(const v8h*)&sX[c0 + 1][g * 8 + 16];                   \
            const v16h bsf = CAT16(ps, qs);                                        \
            AE = __builtin_amdgcn_wmma_f32_16x16x32_f16(false, a1f, false, bf,  (short)0, AE, false, false); \
            AO = __builtin_amdgcn_wmma_f32_16x16x32_f16(false, a0f, false, bf,  (short)0, AO, false, false); \
            AO = __builtin_amdgcn_wmma_f32_16x16x32_f16(false, a2f, false, bsf, (short)0, AO, false, false); \
        }
        DO_SUB(0, accE0, accO0)
        DO_SUB(1, accE1, accO1)
        DO_SUB(2, accE2, accO2)
        DO_SUB(3, accE3, accO3)
#undef DO_SUB

        __syncthreads();
    }

    // Epilogue: C layout lane L -> N = L%16, M = (L/16)*8 + v.
    float bv[8];
#pragma unroll
    for (int v = 0; v < 8; ++v) bv[v] = beff[mb + mw + g * 8 + v];

#define EPI(S, AE, AO)                                                             \
    {                                                                              \
        const int n = nb + nw + (S) * 16 + lm;                                     \
        _Pragma("unroll")                                                          \
        for (int v = 0; v < 8; ++v) {                                              \
            const int e = mb + mw + g * 8 + v;                                     \
            const _Float16 pe = (_Float16)(AE[v] + bv[v]);                         \
            const _Float16 po = (_Float16)(AO[v] + bv[v]);                         \
            const unsigned short ue = *(const unsigned short*)&pe;                 \
            const unsigned short uo = *(const unsigned short*)&po;                 \
            const unsigned pack = (unsigned)ue | ((unsigned)uo << 16);             \
            *(unsigned*)(zh + ((size_t)b * DIMC + e) * LOUT + 2 * n) = pack;       \
        }                                                                          \
    }
    EPI(0, accE0, accO0)
    EPI(1, accE1, accO1)
    EPI(2, accE2, accO2)
    EPI(3, accE3, accO3)
#undef EPI
}

// ---------------------------------------------------------------------------
// Depthwise 17-tap AA filter (commuted past projection) + proj bias, f32 out.
// One block = one (b,e) row segment of 1024 samples, LDS-staged with halo.
// ---------------------------------------------------------------------------
__global__ __launch_bounds__(256)
void dw_proj(const _Float16* __restrict__ zh,
             const float* __restrict__ aa,
             const float* __restrict__ proj_b,
             float* __restrict__ out) {
    __shared__ _Float16 sz[1024 + 16];
    const int row  = blockIdx.y;            // b*DIMC + e
    const int seg  = blockIdx.x * 1024;
    const int tid  = threadIdx.x;
    const size_t base = (size_t)row * LOUT;

    for (int i = tid; i < 1024 + 16; i += 256) {
        const int gl = seg - 8 + i;
        sz[i] = (gl >= 0 && gl < LOUT) ? zh[base + gl] : (_Float16)0.f;
    }
    float k[17];
#pragma unroll
    for (int t = 0; t < 17; ++t) k[t] = aa[t];
    const float pb = proj_b[row & (DIMC - 1)];
    __syncthreads();

    const int j = tid * 4;
    float r[4];
#pragma unroll
    for (int q = 0; q < 4; ++q) {
        float acc = pb;
#pragma unroll
        for (int t = 0; t < 17; ++t) acc += k[t] * (float)sz[j + q + t];
        r[q] = acc;
    }
    float4 v = make_float4(r[0], r[1], r[2], r[3]);
    *(float4*)(out + base + seg + j) = v;
}

// ---------------------------------------------------------------------------
extern "C" void kernel_launch(void* const* d_in, const int* in_sizes, int n_in,
                              void* d_out, int out_size, void* d_ws, size_t ws_size,
                              hipStream_t stream) {
    const float* x       = (const float*)d_in[0];   // [8,512,4096]
    const float* conv_w  = (const float*)d_in[1];   // [512,512,3]
    const float* conv_b  = (const float*)d_in[2];   // [512]
    const float* aa      = (const float*)d_in[3];   // [17]
    const float* proj_w  = (const float*)d_in[4];   // [512,512]
    const float* proj_b  = (const float*)d_in[5];   // [512]
    float* out = (float*)d_out;                     // [8,512,8192]

    char* ws = (char*)d_ws;
    const size_t szA = (size_t)DIMC * DIMC * sizeof(_Float16);   // 512 KiB
    _Float16* A0   = (_Float16*)(ws);
    _Float16* A1   = (_Float16*)(ws + szA);
    _Float16* A2   = (_Float16*)(ws + 2 * szA);
    float*    beff = (float*)   (ws + 3 * szA);
    _Float16* zh   = (_Float16*)(ws + 3 * szA + 2048);           // [8,512,8192] f16

    // 1) fold projection into transposed-conv weights
    prep_Afold<<<dim3(DIMC / 16, DIMC / 16), dim3(16, 16), 0, stream>>>(
        proj_w, conv_w, A0, A1, A2);
    prep_beff<<<2, 256, 0, stream>>>(proj_w, conv_b, beff);

    // 2) fused upsample-conv + projection GEMM (WMMA f16 -> f32)
    up_gemm<<<dim3(DIMC / BM, LIN / BN, NBATCH), 256, 0, stream>>>(
        x, A0, A1, A2, beff, zh);

    // 3) depthwise AA filter + projection bias
    dw_proj<<<dim3(LOUT / 1024, NBATCH * DIMC), 256, 0, stream>>>(
        zh, aa, proj_b, out);
}
